// Involution_23210003267883
// MI455X (gfx1250) — matrix operand order repaired and checked by reading, not compiled, not run on hardware
//
#include <hip/hip_runtime.h>

// ---------------- problem constants ----------------
#define BB    4
#define CC    256
#define HH    128
#define WW    128
#define KK    7
#define PADR  3
#define GG    16
#define GCH   16
#define CRED  64
#define GKK   784          // G*K*K
#define EPSV  1e-5f

// ---------------- tiling ----------------
#define TS    8            // 8x8 spatial tile
#define TP    64           // pixels per tile
#define HALO  (TS + KK - 1) // 14

// LDS strides / offsets (bytes)
#define WM_STR 788                       // wm tile: 64 x 788 f32
#define RS_STR 66                        // r tile : 64 x 66 bf16
#define XH_STR 15                        // halo   : 16 x 14 x 15 f32
#define OFF_RS  ((size_t)64*WM_STR*4)            // 201,728
#define OFF_XS  (OFF_RS + (size_t)64*RS_STR*2)   // 210,176 (16B aligned)
#define XS_BYTES ((size_t)32*TP*4)               // 8,192 per slab buffer
#define OFF_XH  (OFF_XS + 2*XS_BYTES)            // 226,560
#define LDS_TOTAL (OFF_XH + (size_t)GCH*HALO*XH_STR*4)  // 240,000

typedef __attribute__((ext_vector_type(16))) __bf16 v16bf;
typedef __attribute__((ext_vector_type(8)))  float  v8f;
typedef unsigned int uint32x4 __attribute__((ext_vector_type(4)));
typedef int          int32x8  __attribute__((ext_vector_type(8)));
typedef int          int32x4  __attribute__((ext_vector_type(4)));

union BFrag { v16bf v; __bf16 b[16]; unsigned short u[16]; };

static __device__ inline unsigned short bf2u(__bf16 h) {
  union { __bf16 h; unsigned short u; } z; z.h = h; return z.u;
}

// ---- Tensor Data Mover: async 3D tile load  x[cs..cs+31][Y0..Y0+7][X0..X0+7] -> LDS (fp32)
// D# per CDNA5 ISA §8: group0 = control/lds/global addr, group1 = dims+tile, group2 = tensor_dim2.
// This toolchain's builtin is the 6-arg form (extra int32x8 group, then cache policy).
static __device__ inline void tdm_load_slab(const float* gaddr, unsigned lds_off) {
  const unsigned long long ga = (unsigned long long)(uintptr_t)gaddr;
  uint32x4 g0;
  g0[0] = 1u;                                               // count=1 (valid user D#)
  g0[1] = lds_off;                                          // lds_addr (bytes)
  g0[2] = (unsigned)(ga & 0xffffffffu);                     // global_addr[31:0]
  g0[3] = (unsigned)((ga >> 32) & 0x1ffffffu) | (2u << 30); // global_addr[56:32] | type=2
  int32x8 g1;
  g1[0] = (int)(2u << 16);                 // workgroup_mask=0, data_size=2 (4 bytes)
  g1[1] = (int)((unsigned)WW << 16);       // tensor_dim0=128 -> bits[79:48] low half
  g1[2] = (int)((unsigned)HH << 16);       // dim0 hi=0 | tensor_dim1=128 low16
  g1[3] = (int)((unsigned)TS << 16);       // dim1 hi=0 | tile_dim0=8
  g1[4] = (int)((32u << 16) | (unsigned)TS); // tile_dim1=8 | tile_dim2=32
  g1[5] = WW;                              // tensor_dim0_stride = 128 (low 32 of 48)
  g1[6] = (int)((unsigned)(HH*WW) << 16);  // stride0 hi=0 | tensor_dim1_stride=16384 low16
  g1[7] = 0;                               // stride1 hi
  int32x4 g2; g2[0] = CC; g2[1] = 0; g2[2] = 0; g2[3] = 0;  // tensor_dim2=256
  int32x4 g3; g3[0] = 0;  g3[1] = 0; g3[2] = 0; g3[3] = 0;
  int32x8 g4; g4[0] = 0;  g4[1] = 0; g4[2] = 0; g4[3] = 0;
              g4[4] = 0;  g4[5] = 0; g4[6] = 0; g4[7] = 0;
  __builtin_amdgcn_tensor_load_to_lds(g0, g1, g2, g3, g4, 0);
}

extern "C" __global__ __launch_bounds__(256)
void involution_fused(const float* __restrict__ x,
                      const float* __restrict__ w_reduce,
                      const float* __restrict__ b_reduce,
                      const float* __restrict__ bn_gamma,
                      const float* __restrict__ bn_beta,
                      const float* __restrict__ bn_mean,
                      const float* __restrict__ bn_var,
                      const float* __restrict__ w_span,
                      const float* __restrict__ b_span,
                      float* __restrict__ out)
{
  extern __shared__ char smem[];
  float*          wm_s = (float*)smem;                       // 64*788 f32
  unsigned short* r_s  = (unsigned short*)(smem + OFF_RS);   // 64*66 bf16
  float*          xs0  = (float*)(smem + OFF_XS);            // double-buffered fp32 x slab
  float*          xs1  = (float*)(smem + OFF_XS + XS_BYTES);
  float*          xh   = (float*)(smem + OFF_XH);            // halo tile

  const int tid  = threadIdx.x;
  const int lane = tid & 31;
  const int wid  = tid >> 5;

  const int tilesX = WW / TS;                // 16
  const int b  = blockIdx.x / (tilesX * tilesX);
  const int tr = blockIdx.x % (tilesX * tilesX);
  const int Y0 = (tr / tilesX) * TS;
  const int X0 = (tr % tilesX) * TS;

  // WMMA lane decomposition (per CDNA5 ISA VGPR layouts)
  const int laneM     = lane & 15;
  const int laneHiK8  = (lane & 16) ? 8 : 0;  // A-matrix K base for upper half-wave
  const int laneHiK16 = (lane & 16);          // unused; B also takes its K base from laneHiK8
  const int laneHiM8  = (lane & 16) ? 8 : 0;  // C/D M offset for upper half-wave

  // ---------------- Phase A: r = ReLU(BN(x . WrT + br)),  M=64 N=64 K=256 ----------------
  const int ntA  = wid & 3;
  const int mtA0 = wid >> 2;
  const int mtA1 = mtA0 + 2;

  v8f acc0 = {0.f,0.f,0.f,0.f,0.f,0.f,0.f,0.f};
  v8f acc1 = {0.f,0.f,0.f,0.f,0.f,0.f,0.f,0.f};

  const float* slab_base = x + (((size_t)b*CC*HH + Y0) * WW + X0);
  // prime slab 0 via Tensor Data Mover (wave 0 issues; TENSORcnt tracks)
  if (wid == 0)
    tdm_load_slab(slab_base, (unsigned)(uintptr_t)xs0);

  for (int ks = 0; ks < CC / 32; ++ks) {
    if (wid == 0) __builtin_amdgcn_s_wait_tensorcnt(0);   // slab ks landed in LDS
    __syncthreads();
    const float* xs = (ks & 1) ? xs1 : xs0;
    if (wid == 0 && ks + 1 < CC / 32)                     // overlap next DMA with WMMA
      tdm_load_slab(slab_base + (size_t)(ks + 1) * 32 * HH * WW,
                    (unsigned)(uintptr_t)(((ks + 1) & 1) ? xs1 : xs0));

    BFrag af0, af1, bf;
    #pragma unroll
    for (int e = 0; e < 16; ++e) {
      const int kk = laneHiK8 + (e & 7) + ((e & 8) << 1);   // 0..31 within slab
      af0.b[e] = (__bf16)xs[kk*TP + (mtA0*16 + laneM)];     // TDM layout: [c][y][x]
      af1.b[e] = (__bf16)xs[kk*TP + (mtA1*16 + laneM)];
      const int kb = ks*32 + laneHiK8 + (e & 7) + ((e & 8) << 1); // global K (channel)
      bf.b[e]  = (__bf16)w_reduce[(ntA*16 + laneM)*CC + kb];// B[k][n] = Wr[n][k]
    }
    acc0 = __builtin_amdgcn_wmma_f32_16x16x32_bf16(false, af0.v, false, bf.v, (short)0, acc0, false, false);
    acc1 = __builtin_amdgcn_wmma_f32_16x16x32_bf16(false, af1.v, false, bf.v, (short)0, acc1, false, false);
    __syncthreads();   // everyone done reading xs[ks&1] before it is re-targeted
  }

  // epilogue: bias + BN(eval) + ReLU, write r (bf16) to LDS
  {
    const int o   = ntA*16 + laneM;
    const float a  = bn_gamma[o] * rsqrtf(bn_var[o] + EPSV);
    const float sh = b_reduce[o] - bn_mean[o];
    const float bt = bn_beta[o];
    #pragma unroll
    for (int r = 0; r < 8; ++r) {
      float v0 = (acc0[r] + sh) * a + bt;  v0 = v0 > 0.f ? v0 : 0.f;
      float v1 = (acc1[r] + sh) * a + bt;  v1 = v1 > 0.f ? v1 : 0.f;
      r_s[(mtA0*16 + r + laneHiM8)*RS_STR + o] = bf2u((__bf16)v0);
      r_s[(mtA1*16 + r + laneHiM8)*RS_STR + o] = bf2u((__bf16)v1);
    }
  }
  __syncthreads();

  // ---------------- Phase B: wm = r . WsT + bs,  M=64 N=784 K=64 ----------------
  for (int t = wid; t < 196; t += 8) {
    const int mt = t & 3;
    const int nt = t >> 2;              // 0..48
    if (t + 8 < 196)
      __builtin_prefetch(&w_span[((t + 8) >> 2) * 16 * CRED], 0, 0);
    v8f acc = {0.f,0.f,0.f,0.f,0.f,0.f,0.f,0.f};
    #pragma unroll
    for (int ks = 0; ks < 2; ++ks) {
      BFrag af, bf;
      #pragma unroll
      for (int e = 0; e < 16; ++e) {
        const int kk = ks*32 + laneHiK8 + (e & 7) + ((e & 8) << 1);
        af.u[e] = r_s[(mt*16 + laneM)*RS_STR + kk];
        const int kb = ks*32 + laneHiK8 + (e & 7) + ((e & 8) << 1);
        bf.b[e] = (__bf16)w_span[(nt*16 + laneM)*CRED + kb];  // B[k][n] = Ws[n][k]
      }
      acc = __builtin_amdgcn_wmma_f32_16x16x32_bf16(false, af.v, false, bf.v, (short)0, acc, false, false);
    }
    const int n  = nt*16 + laneM;
    const float bs = b_span[n];
    #pragma unroll
    for (int r = 0; r < 8; ++r)
      wm_s[(mt*16 + r + laneHiM8)*WM_STR + n] = acc[r] + bs;
  }
  __syncthreads();

  // ---------------- Phase C: dynamic depthwise conv (16 groups) ----------------
  const int p  = tid & 63;      // pixel within tile
  const int c4 = tid >> 6;      // 0..3 -> 4 channels each
  const int py = p >> 3, px = p & 7;

  for (int g = 0; g < GG; ++g) {
    // zero-padded halo for this group's 16 channels (second read of x: L2-resident)
    for (int idx = tid; idx < GCH*HALO*HALO; idx += 256) {
      const int c  = idx / (HALO*HALO);
      const int rm = idx % (HALO*HALO);
      const int hy = rm / HALO, hx = rm % HALO;
      const int gy = Y0 + hy - PADR, gx = X0 + hx - PADR;
      float v = 0.f;
      if (gy >= 0 && gy < HH && gx >= 0 && gx < WW)
        v = x[(((size_t)(b*CC + g*GCH + c)) * HH + gy) * WW + gx];
      xh[(c*HALO + hy)*XH_STR + hx] = v;
    }
    __syncthreads();

    float acc[4] = {0.f, 0.f, 0.f, 0.f};
    const float* wmp = &wm_s[p*WM_STR + g*49];
    #pragma unroll
    for (int i = 0; i < KK; ++i) {
      #pragma unroll
      for (int j = 0; j < KK; ++j) {
        const float w = wmp[i*KK + j];              // broadcast across the 16 channels
        #pragma unroll
        for (int cc = 0; cc < 4; ++cc) {
          const int c = c4*4 + cc;
          acc[cc] = fmaf(w, xh[(c*HALO + (py + i))*XH_STR + (px + j)], acc[cc]);
        }
      }
    }
    #pragma unroll
    for (int cc = 0; cc < 4; ++cc) {
      const int c = c4*4 + cc;
      out[(((size_t)(b*CC + g*GCH + c)) * HH + (Y0 + py)) * WW + (X0 + px)] = acc[cc];
    }
    __syncthreads();   // before next group's halo overwrites xh
  }
}

extern "C" void kernel_launch(void* const* d_in, const int* in_sizes, int n_in,
                              void* d_out, int out_size, void* d_ws, size_t ws_size,
                              hipStream_t stream) {
  const float* x        = (const float*)d_in[0];
  const float* w_reduce = (const float*)d_in[1];
  const float* b_reduce = (const float*)d_in[2];
  const float* bn_gamma = (const float*)d_in[3];
  const float* bn_beta  = (const float*)d_in[4];
  const float* bn_mean  = (const float*)d_in[5];
  const float* bn_var   = (const float*)d_in[6];
  const float* w_span   = (const float*)d_in[7];
  const float* b_span   = (const float*)d_in[8];
  float* out = (float*)d_out;

  const dim3 grid(BB * (HH/TS) * (WW/TS));  // 1024 workgroups
  involution_fused<<<grid, 256, LDS_TOTAL, stream>>>(x, w_reduce, b_reduce, bn_gamma, bn_beta,
                                                     bn_mean, bn_var, w_span, b_span, out);
}
